// HybridHead_44968307589405
// MI455X (gfx1250) — matrix operand order repaired and checked by reading, not compiled, not run on hardware
//
#include <hip/hip_runtime.h>

typedef __bf16 bf16;
typedef __attribute__((ext_vector_type(16))) __bf16 v16bf;
typedef __attribute__((ext_vector_type(8)))  float  v8f;

#define D_DIM     2048
#define B_ROWS    128
#define M_ROWS    65536
#define C_CLASSES 8192
#define N2        256
#define T_SPCL_INV 20.0f
#define T_TMP_INV  10.0f

// ---- workspace layout (bytes, 256B-aligned chunks) ----
#define SIMACC_OFF 0u                     // 8192*128*4 = 4194304
#define COUNTS_OFF 4194304u               // 8192*4    = 32768
#define XHI_OFF    4227072u               // 128*2048*2 = 524288
#define XLO_OFF    4751360u
#define FCHI_OFF   5275648u               // 256*2048*2 = 1048576
#define FCLO_OFF   6324224u
#define G_OFF      7372800u               // 256*256*4 = 262144
#define LOSS_OFF   7634944u               // scalar accumulators

// ---------------- helpers ----------------
__device__ __forceinline__ v16bf load32B(const bf16* p) {  // 16 contiguous bf16
  union { v16bf v; unsigned long long u[4]; } f;
  const unsigned long long* q = (const unsigned long long*)p;
  f.u[0] = q[0]; f.u[1] = q[1]; f.u[2] = q[2]; f.u[3] = q[3];
  return f.v;
}
__device__ __forceinline__ v16bf load2x16B(const bf16* p0, const bf16* p1) { // two 8-bf16 runs
  union { v16bf v; unsigned long long u[4]; } f;
  const unsigned long long* a = (const unsigned long long*)p0;
  const unsigned long long* b = (const unsigned long long*)p1;
  f.u[0] = a[0]; f.u[1] = a[1]; f.u[2] = b[0]; f.u[3] = b[1];
  return f.v;
}
__device__ __forceinline__ v8f wmma_bf16(v16bf a, v16bf b, v8f c) {
  return __builtin_amdgcn_wmma_f32_16x16x32_bf16(false, a, false, b, (short)0, c, false, false);
}
__device__ __forceinline__ void split_bf16(float x, bf16& hi, bf16& lo) {
  bf16 h = (bf16)x;
  hi = h;
  lo = (bf16)(x - (float)h);
}
// native fp32 atomic add (global_atomic_add_f32, no CAS loop)
__device__ __forceinline__ void atomAddF(float* p, float v) {
  unsafeAtomicAdd(p, v);
}

// ---------------- prep: normalize inputs, split to bf16 hi/lo ----------------
__global__ __launch_bounds__(256) void prep_x(const float* __restrict__ in,
                                              bf16* __restrict__ xhi,
                                              bf16* __restrict__ xlo) {
  __shared__ float red[256];
  const int row = blockIdx.x, tid = threadIdx.x;
  const float* src = in + (size_t)row * D_DIM;
  float v[8], ssq = 0.f;
#pragma unroll
  for (int i = 0; i < 8; ++i) { v[i] = src[tid + i * 256]; ssq += v[i] * v[i]; }
  red[tid] = ssq; __syncthreads();
  for (int o = 128; o > 0; o >>= 1) { if (tid < o) red[tid] += red[tid + o]; __syncthreads(); }
  const float rinv = 1.0f / sqrtf(red[0]);
#pragma unroll
  for (int i = 0; i < 8; ++i) {
    bf16 h, l; split_bf16(v[i] * rinv, h, l);
    xhi[(size_t)row * D_DIM + tid + i * 256] = h;
    xlo[(size_t)row * D_DIM + tid + i * 256] = l;
  }
}

// ---------------- prep: feats_aug [B,2,D] -> f [2B,D] bf16 hi/lo ----------------
__global__ __launch_bounds__(256) void prep_f(const float* __restrict__ fa,
                                              bf16* __restrict__ fhi,
                                              bf16* __restrict__ flo) {
  const int row = blockIdx.x, tid = threadIdx.x;
  const float* src = fa + (size_t)((row & 127) * 2 + (row >> 7)) * D_DIM;
  bf16* dh = fhi + (size_t)row * D_DIM;
  bf16* dl = flo + (size_t)row * D_DIM;
#pragma unroll
  for (int i = 0; i < 8; ++i) {
    bf16 h, l; split_bf16(src[tid + i * 256], h, l);
    dh[tid + i * 256] = h;
    dl[tid + i * 256] = l;
  }
}

// ---------------- per-class member counts ----------------
__global__ __launch_bounds__(256) void count_k(const int* __restrict__ ml,
                                               int* __restrict__ counts) {
  const int m = blockIdx.x * 256 + threadIdx.x;
  atomicAdd(&counts[ml[m]], 1);
}

// ---------------- big GEMM + fused segment-sum ----------------
// S = X[128,2048] @ F^T, bf16 hi/lo split (3 WMMA terms), atomically scattered
// into sim_acc[class, b]. One workgroup = 64 M-columns, 8 waves = all 128 B rows.
__global__ __launch_bounds__(256) void spcl_gemm(const float* __restrict__ F,
                                                 const bf16* __restrict__ xhi,
                                                 const bf16* __restrict__ xlo,
                                                 const int*  __restrict__ ml,
                                                 float* __restrict__ simacc) {
  __shared__ bf16 Bh[64 * 32];
  __shared__ bf16 Bl[64 * 32];
  const int tid = threadIdx.x;
  const int wid = tid >> 5, lane = tid & 31;
  const int r = lane & 15, h = lane >> 4;
  const size_t mbase = (size_t)blockIdx.x * 64;

  // staging assignment: each thread owns one 8-float run of the 64x32 tile
  const int sml = tid >> 2;          // 0..63 (M row in tile)
  const int skk = (tid & 3) * 8;     // 0,8,16,24 (K offset)
  const float* fsrc = F + (mbase + sml) * (size_t)D_DIM + skk;
  bf16* bh_st = Bh + sml * 32 + skk;
  bf16* bl_st = Bl + sml * 32 + skk;

  const bf16* arow_hi = xhi + (size_t)(wid * 16 + r) * D_DIM + 8 * h;
  const bf16* arow_lo = xlo + (size_t)(wid * 16 + r) * D_DIM + 8 * h;

  v8f acc[4] = {};

  for (int kt = 0; kt < 64; ++kt) {
    const int k0 = kt * 32;
    // ---- cooperative stage + fp32 -> bf16 hi/lo convert (done once per WG) ----
    const float4* p = (const float4*)(fsrc + k0);
    float4 f0 = p[0], f1 = p[1];
    if (kt < 63) __builtin_prefetch(fsrc + k0 + 32, 0, 3);   // global_prefetch_b8
    float xs[8] = {f0.x, f0.y, f0.z, f0.w, f1.x, f1.y, f1.z, f1.w};
#pragma unroll
    for (int i = 0; i < 8; ++i) {
      bf16 hh, ll; split_bf16(xs[i], hh, ll);
      bh_st[i] = hh; bl_st[i] = ll;
    }
    __syncthreads();

    // ---- A fragments (global, L2-hot) ----
    v16bf ahi = load2x16B(arow_hi + k0, arow_hi + k0 + 16);
    v16bf alo = load2x16B(arow_lo + k0, arow_lo + k0 + 16);
#pragma unroll
    for (int nt = 0; nt < 4; ++nt) {
      const bf16* bp = Bh + (nt * 16 + r) * 32 + 8 * h;
      const bf16* lp = Bl + (nt * 16 + r) * 32 + 8 * h;
      v16bf bhi = load2x16B(bp, bp + 16);
      v16bf blo = load2x16B(lp, lp + 16);
      acc[nt] = wmma_bf16(ahi, bhi, acc[nt]);   // hi*hi
      acc[nt] = wmma_bf16(ahi, blo, acc[nt]);   // hi*lo
      acc[nt] = wmma_bf16(alo, bhi, acc[nt]);   // lo*hi
    }
    __syncthreads();
  }

  // ---- epilogue: fused segment-sum (sims never materialized) ----
#pragma unroll
  for (int nt = 0; nt < 4; ++nt) {
    const size_t m = mbase + nt * 16 + r;
    const int c = ml[m];
    float* dst = simacc + (size_t)c * B_ROWS + wid * 16 + 8 * h;
#pragma unroll
    for (int e = 0; e < 8; ++e) atomAddF(dst + e, acc[nt][e]);
  }
}

// ---------------- SPCL masked softmax + loss ----------------
__global__ __launch_bounds__(256) void spcl_finish(const float* __restrict__ simacc,
                                                   const int* __restrict__ counts,
                                                   const int* __restrict__ idx,
                                                   const int* __restrict__ ml,
                                                   float* __restrict__ loss) {
  __shared__ float rsum[256];
  __shared__ float rtgt[256];
  const int b = blockIdx.x, tid = threadIdx.x;
  const int tgt = ml[idx[b]];
  float s = 0.f, et = 0.f;
  for (int c = tid; c < C_CLASSES; c += 256) {
    const int n = counts[c];
    if (n > 0) {
      const float sim = simacc[(size_t)c * B_ROWS + b] * T_SPCL_INV / (float)n;
      const float e = expf(sim);      // reference has no max-shift
      s += e;
      if (c == tgt) et = e;
    }
  }
  rsum[tid] = s; rtgt[tid] = et; __syncthreads();
  for (int o = 128; o > 0; o >>= 1) {
    if (tid < o) { rsum[tid] += rsum[tid + o]; rtgt[tid] += rtgt[tid + o]; }
    __syncthreads();
  }
  if (tid == 0) {
    const float ms = rtgt[0] / (rsum[0] + 1e-6f);
    atomAddF(loss, -logf(ms + 1e-6f) * (1.0f / 128.0f));
  }
}

// ---------------- contrastive GEMM: G = f @ f^T (256x256) ----------------
__global__ __launch_bounds__(256) void contr_gemm(const bf16* __restrict__ fhi,
                                                  const bf16* __restrict__ flo,
                                                  float* __restrict__ G) {
  const int tid = threadIdx.x, wid = tid >> 5, lane = tid & 31;
  const int r = lane & 15, h = lane >> 4;
  const int t = blockIdx.x * 8 + wid;     // 256 tiles of 16x16
  const int it = t >> 4, jt = t & 15;
  const bf16* ah = fhi + (size_t)(it * 16 + r) * D_DIM + 8 * h;
  const bf16* al = flo + (size_t)(it * 16 + r) * D_DIM + 8 * h;
  const bf16* bh = fhi + (size_t)(jt * 16 + r) * D_DIM + 8 * h;
  const bf16* bl = flo + (size_t)(jt * 16 + r) * D_DIM + 8 * h;
  v8f acc = {};
  for (int k0 = 0; k0 < D_DIM; k0 += 32) {
    v16bf A   = load2x16B(ah + k0, ah + k0 + 16);
    v16bf Alv = load2x16B(al + k0, al + k0 + 16);
    v16bf Bv  = load2x16B(bh + k0, bh + k0 + 16);
    v16bf Blv = load2x16B(bl + k0, bl + k0 + 16);
    acc = wmma_bf16(A, Bv, acc);
    acc = wmma_bf16(A, Blv, acc);
    acc = wmma_bf16(Alv, Bv, acc);
  }
#pragma unroll
  for (int e = 0; e < 8; ++e)
    G[(size_t)(it * 16 + 8 * h + e) * N2 + jt * 16 + r] = acc[e];
}

// ---------------- contrastive loss (one row per workgroup) ----------------
__global__ __launch_bounds__(256) void contr_finish(const float* __restrict__ G,
                                                    const int* __restrict__ label,
                                                    float* __restrict__ loss) {
  __shared__ float red[256];
  __shared__ float b0, b1;
  const int i = blockIdx.x, j = threadIdx.x;
  const bool valid = (j != i);
  const float Lj = valid ? G[(size_t)i * N2 + j] * T_TMP_INV : -3.0e38f;
  const bool pos = valid && (label[i & 127] == label[j & 127]);

  // row max (over j != i)
  red[j] = Lj; __syncthreads();
  for (int o = 128; o > 0; o >>= 1) { if (j < o) red[j] = fmaxf(red[j], red[j + o]); __syncthreads(); }
  if (j == 0) b0 = red[0];
  __syncthreads();
  const float m = b0;
  const float E = valid ? expf(Lj - m) : 0.f;

  // negative sum
  red[j] = (valid && !pos) ? E : 0.f; __syncthreads();
  for (int o = 128; o > 0; o >>= 1) { if (j < o) red[j] += red[j + o]; __syncthreads(); }
  if (j == 0) b1 = red[0];
  __syncthreads();
  const float negs = b1;

  // positive count
  red[j] = pos ? 1.f : 0.f; __syncthreads();
  for (int o = 128; o > 0; o >>= 1) { if (j < o) red[j] += red[j + o]; __syncthreads(); }
  if (j == 0) b0 = red[0];
  __syncthreads();
  const float pcnt = b0;

  // sum of per-positive losses
  red[j] = pos ? (logf(E + negs) + m - Lj) : 0.f; __syncthreads();
  for (int o = 128; o > 0; o >>= 1) { if (j < o) red[j] += red[j + o]; __syncthreads(); }
  if (j == 0) atomAddF(loss, (red[0] / pcnt) * (1.0f / 256.0f));
}

__global__ void write_out(const float* __restrict__ loss, float* __restrict__ out) {
  out[0] = loss[0];
}

extern "C" void kernel_launch(void* const* d_in, const int* in_sizes, int n_in,
                              void* d_out, int out_size, void* d_ws, size_t ws_size,
                              hipStream_t stream) {
  (void)in_sizes; (void)n_in; (void)out_size; (void)ws_size;
  const float* inputs     = (const float*)d_in[0];
  const float* feats_aug  = (const float*)d_in[1];
  const float* features   = (const float*)d_in[2];
  const int*   idx        = (const int*)d_in[3];
  const int*   label      = (const int*)d_in[4];
  const int*   mem_labels = (const int*)d_in[5];

  char* ws = (char*)d_ws;
  float* simacc = (float*)(ws + SIMACC_OFF);
  int*   counts = (int*)(ws + COUNTS_OFF);
  bf16*  xhi    = (bf16*)(ws + XHI_OFF);
  bf16*  xlo    = (bf16*)(ws + XLO_OFF);
  bf16*  fchi   = (bf16*)(ws + FCHI_OFF);
  bf16*  fclo   = (bf16*)(ws + FCLO_OFF);
  float* G      = (float*)(ws + G_OFF);
  float* lossac = (float*)(ws + LOSS_OFF);

  hipMemsetAsync(ws + SIMACC_OFF, 0, 4194304u + 32768u, stream);  // sim_acc + counts
  hipMemsetAsync(ws + LOSS_OFF, 0, 16, stream);

  prep_x      <<<128, 256, 0, stream>>>(inputs, xhi, xlo);
  prep_f      <<<256, 256, 0, stream>>>(feats_aug, fchi, fclo);
  count_k     <<<M_ROWS / 256, 256, 0, stream>>>(mem_labels, counts);
  spcl_gemm   <<<M_ROWS / 64, 256, 0, stream>>>(features, xhi, xlo, mem_labels, simacc);
  spcl_finish <<<B_ROWS, 256, 0, stream>>>(simacc, counts, idx, mem_labels, lossac);
  contr_gemm  <<<32, 256, 0, stream>>>(fchi, fclo, G);
  contr_finish<<<N2, 256, 0, stream>>>(G, label, lossac);
  write_out   <<<1, 1, 0, stream>>>(lossac, (float*)d_out);
}
